// hyperedge_attn_57337813402298
// MI455X (gfx1250) — matrix operand order repaired and checked by reading, not compiled, not run on hardware
//
#include <hip/hip_runtime.h>

typedef __attribute__((ext_vector_type(16))) _Float16 v16h;
typedef __attribute__((ext_vector_type(8)))  float    v8f;

#define B_   32
#define S_   1024
#define F_   128
#define HID_ 64
#define E_   32
#define K_   32
#define M_   (B_ * S_)   /* 32768 rows for projections */
#define NPROJ 128        /* [W | Wr] packed */
#define ROWT 2           /* row tiles per wave in industry attention */
#define AW_  (S_ / 32)   /* adj bitmask words per row */

// ---- CDNA5 16-bit A-matrix 16x32 lane/half -> K mapping (contiguous, matches B) ----
// lanes 0-15: halves 0-15 -> K=0..15
// lanes 16-31: halves 0-15 -> K=16..31
__device__ __forceinline__ int akmap(int h, int lane) {
  int base = (lane >> 4) << 4;
  return base + h;
}
__device__ __forceinline__ float lrelu(float z) { return z > 0.f ? z : 0.01f * z; }
__device__ __forceinline__ float eluf(float z)  { return z > 0.f ? z : __expf(z) - 1.f; }

// ---------------------------------------------------------------- zero fill
__global__ void zero_f4_kernel(float4* __restrict__ p, long n4) {
  long i = (long)blockIdx.x * blockDim.x + threadIdx.x;
  long stride = (long)gridDim.x * blockDim.x;
  for (; i < n4; i += stride) p[i] = make_float4(0.f, 0.f, 0.f, 0.f);
}

// ---------------------------------------------------------------- f32 -> f16
__global__ void cvt_f16_kernel(const float* __restrict__ x, _Float16* __restrict__ xh, int n) {
  int i = blockIdx.x * blockDim.x + threadIdx.x;
  if (i < n) xh[i] = (_Float16)x[i];
}

// ------------------------------------------- adj (f32 0/1) -> bit matrix
// adjbits[row][w] bit j = (adj[row][w*32+j] > 0)
__global__ void pack_adj_kernel(const float* __restrict__ adj, unsigned* __restrict__ adjbits) {
  int i = blockIdx.x * blockDim.x + threadIdx.x;  // word index
  if (i >= S_ * AW_) return;
  int row = i / AW_, wj = i % AW_;
  const float* a = adj + (long)row * S_ + wj * 32;
  unsigned w = 0u;
#pragma unroll
  for (int bit = 0; bit < 32; ++bit)
    if (a[bit] > 0.f) w |= (1u << bit);
  adjbits[i] = w;
}

// ----------------------------------- pack [W|Wr] transposed f16 + bias concat
// WcatT layout: [NPROJ][F_]  (column-major weights -> contiguous B fragments)
__global__ void pack_w_kernel(const float* __restrict__ W, const float* __restrict__ Wr,
                              const float* __restrict__ b, const float* __restrict__ br,
                              _Float16* __restrict__ WcatT, float* __restrict__ bcat) {
  int i = blockIdx.x * blockDim.x + threadIdx.x;
  if (i < F_ * NPROJ) {
    int n = i / F_, k = i % F_;
    float v = (n < HID_) ? W[k * HID_ + n] : Wr[k * HID_ + (n - HID_)];
    WcatT[i] = (_Float16)v;
  }
  if (i < NPROJ) bcat[i] = (i < HID_) ? b[i] : br[i - HID_];
}

// -------------------------------------- WMMA projection GEMM: proj = x@W + b
// grid (M_/16, NPROJ/16), block 32 (one wave / 16x16 C tile)
// Writes: proj (f32, [M_][NPROJ])  and  ftsT (f16, [B_][HID_][S_]) for n<HID_
__global__ void proj_gemm_kernel(const _Float16* __restrict__ xh,
                                 const _Float16* __restrict__ WcatT,
                                 const float* __restrict__ bcat,
                                 float* __restrict__ proj, _Float16* __restrict__ ftsT) {
  int l  = threadIdx.x;
  int i0 = blockIdx.x * 16;
  int n0 = blockIdx.y * 16;
  int row  = i0 + (l & 15);
  int col  = n0 + (l & 15);
  int kb16 = (l >> 4) << 4;  // B-fragment K base: lanes 0-15 -> K 0..15, 16-31 -> 16..31
  v8f acc = {};
  for (int kb = 0; kb < F_; kb += 32) {
    v16h a, bm;
#pragma unroll
    for (int h = 0; h < 16; ++h) a[h] = xh[row * F_ + kb + akmap(h, l)];
#pragma unroll
    for (int h = 0; h < 16; ++h) bm[h] = WcatT[col * F_ + kb + kb16 + h];  // contiguous
    acc = __builtin_amdgcn_wmma_f32_16x16x32_f16(false, a, false, bm, (short)0, acc,
                                                 false, false);
  }
  float bias = bcat[col];
#pragma unroll
  for (int g = 0; g < 8; ++g) {
    int r = i0 + g + ((l >> 4) << 3);
    float v = acc[g] + bias;
    proj[(long)r * NPROJ + col] = v;
    if (n0 < HID_) {  // f16 transposed copy of fts only (B-side of attention)
      int bb = r >> 10, sidx = r & (S_ - 1);
      ftsT[((long)bb * HID_ + col) * S_ + sidx] = (_Float16)v;
    }
  }
}

// ----------------------------------------------- f1/f2 row dots (HID=64 dot)
__global__ void f1f2_kernel(const float* __restrict__ proj,
                            const float* __restrict__ a1, const float* __restrict__ a1b,
                            const float* __restrict__ a2, const float* __restrict__ a2b,
                            float* __restrict__ f1, float* __restrict__ f2) {
  int r = blockIdx.x * blockDim.x + threadIdx.x;
  if (r >= M_) return;
  const float* fts = proj + (long)r * NPROJ;
  float s1 = a1b[0], s2 = a2b[0];
#pragma unroll 8
  for (int k = 0; k < HID_; ++k) {
    float v = fts[k];
    s1 += v * a1[k];
    s2 += v * a2[k];
  }
  f1[r] = s1;
  f2[r] = s2;
}

// ------------------------------------- industry branch: fused masked softmax
// grid (S_/(16*ROWT), B_), block 32. One wave owns ROWT 16-row tiles and
// reuses the B fragments (ftsT) across them. adj comes in as a bit matrix
// (one u32 per lane per row per j-tile instead of 16 f32 loads).
__global__ void industry_attn_kernel(const float* __restrict__ proj,
                                     const _Float16* __restrict__ ftsT,
                                     const float* __restrict__ f1,
                                     const float* __restrict__ f2,
                                     const unsigned* __restrict__ adjbits,
                                     float* __restrict__ out) {
  __shared__ float f1s[S_];
  int l  = threadIdx.x;
  int b  = blockIdx.y;
  int i0 = blockIdx.x * (16 * ROWT);
  const float* f1b = f1 + (long)b * S_;
  float lmax = -1e30f;
  for (int j = l; j < S_; j += 32) {
    float v = f1b[j];
    f1s[j] = v;
    lmax = fmaxf(lmax, v);
  }
#pragma unroll
  for (int sh = 16; sh >= 1; sh >>= 1) lmax = fmaxf(lmax, __shfl_xor(lmax, sh));
  __syncthreads();

  int c    = l & 15;
  int kb16 = (l >> 4) << 4;
  float f2r[ROWT], mrow[ROWT], psum[ROWT];
  const unsigned* abrow[ROWT];
#pragma unroll
  for (int rt = 0; rt < ROWT; ++rt) {
    int rrow  = i0 + rt * 16 + c;
    f2r[rt]   = f2[(long)b * S_ + rrow];
    mrow[rt]  = lrelu(f2r[rt] + lmax);  // upper bound over j -> exp() <= 1
    abrow[rt] = adjbits + (long)rrow * AW_;
    psum[rt]  = 0.f;
  }
  const _Float16* ftT = ftsT + (long)b * HID_ * S_;

  v8f acc[ROWT][4];
#pragma unroll
  for (int rt = 0; rt < ROWT; ++rt)
#pragma unroll
    for (int nt = 0; nt < 4; ++nt) acc[rt][nt] = (v8f){};

  for (int j0 = 0; j0 < S_; j0 += 32) {
    if (j0 + 32 < S_) {  // gfx1250 global_prefetch_b8 for next streamed tile
      __builtin_prefetch(&ftT[(long)(c)      * S_ + j0 + 32 + kb16], 0, 3);
      __builtin_prefetch(&ftT[(long)(16 + c) * S_ + j0 + 32 + kb16], 0, 3);
      __builtin_prefetch(&ftT[(long)(32 + c) * S_ + j0 + 32 + kb16], 0, 3);
      __builtin_prefetch(&ftT[(long)(48 + c) * S_ + j0 + 32 + kb16], 0, 3);
    }
    v16h b0, b1, b2, b3;  // shared across ROWT row tiles
#pragma unroll
    for (int h = 0; h < 16; ++h) {
      int j = j0 + kb16 + h;  // contiguous 32B per lane -> b128 loads
      b0[h] = ftT[(long)(c)      * S_ + j];
      b1[h] = ftT[(long)(16 + c) * S_ + j];
      b2[h] = ftT[(long)(32 + c) * S_ + j];
      b3[h] = ftT[(long)(48 + c) * S_ + j];
    }
#pragma unroll
    for (int rt = 0; rt < ROWT; ++rt) {
      unsigned mw = abrow[rt][j0 >> 5];  // 32 mask bits for this j-tile
      v16h a;
      float ps = 0.f;
#pragma unroll
      for (int h = 0; h < 16; ++h) {
        int   jb = akmap(h, l);
        float p  = ((mw >> jb) & 1u)
                       ? __expf(lrelu(f2r[rt] + f1s[j0 + jb]) - mrow[rt])
                       : 0.f;
        a[h] = (_Float16)p;
        ps += p;
      }
      psum[rt] += ps;
      acc[rt][0] = __builtin_amdgcn_wmma_f32_16x16x32_f16(false, a, false, b0, (short)0, acc[rt][0], false, false);
      acc[rt][1] = __builtin_amdgcn_wmma_f32_16x16x32_f16(false, a, false, b1, (short)0, acc[rt][1], false, false);
      acc[rt][2] = __builtin_amdgcn_wmma_f32_16x16x32_f16(false, a, false, b2, (short)0, acc[rt][2], false, false);
      acc[rt][3] = __builtin_amdgcn_wmma_f32_16x16x32_f16(false, a, false, b3, (short)0, acc[rt][3], false, false);
    }
  }
#pragma unroll
  for (int rt = 0; rt < ROWT; ++rt) {
    float rowsum = psum[rt] + __shfl_xor(psum[rt], 16);  // full row sum, row (l&15)
#pragma unroll
    for (int g = 0; g < 8; ++g) {
      int   rM = g + ((l >> 4) << 3);
      float rs = __shfl(rowsum, rM);
      long  ridx = (long)b * S_ + i0 + rt * 16 + rM;
      const float* resid = proj + ridx * NPROJ + HID_;
      float* o = out + ridx * HID_;
      o[c]      = eluf(acc[rt][0][g] / rs + resid[c]);
      o[16 + c] = eluf(acc[rt][1][g] / rs + resid[16 + c]);
      o[32 + c] = eluf(acc[rt][2][g] / rs + resid[32 + c]);
      o[48 + c] = eluf(acc[rt][3][g] / rs + resid[48 + c]);
    }
  }
}

// ------------------------ hyperedge branch: 32-node head + scatter + maxpool
// grid (E_, B_), block 64 (2 waves, each one 16-row tile; K=32 in one WMMA)
__global__ void hyper_attn_kernel(const float* __restrict__ proj,
                                  const _Float16* __restrict__ ftsT,
                                  const float* __restrict__ f1,
                                  const float* __restrict__ f2,
                                  float* __restrict__ out_all,
                                  float* __restrict__ out_pool) {
  __shared__ float f1s[K_];
  __shared__ float sh_after[K_][HID_];
  int t = threadIdx.x;
  int w = t >> 5;
  int l = t & 31;
  int e = blockIdx.x;
  int b = blockIdx.y;
  long base = (long)b * S_ + e * K_;  // row of member 0 in proj/f1/f2
  if (t < K_) f1s[t] = f1[base + t];
  __syncthreads();

  float lmax = f1s[l];
#pragma unroll
  for (int sh = 16; sh >= 1; sh >>= 1) lmax = fmaxf(lmax, __shfl_xor(lmax, sh));

  int   i0   = w * 16;
  int   c    = l & 15;
  int   kb16 = (l >> 4) << 4;
  float f2r  = f2[base + i0 + c];
  float mrow = lrelu(f2r + lmax);

  v16h a;
  float psum = 0.f;
#pragma unroll
  for (int h = 0; h < 16; ++h) {
    float p = __expf(lrelu(f2r + f1s[akmap(h, l)]) - mrow);
    a[h] = (_Float16)p;
    psum += p;
  }
  const _Float16* ftT = ftsT + (long)b * HID_ * S_;
  int jbase = e * K_ + kb16;
  v16h b0, b1, b2, b3;
#pragma unroll
  for (int h = 0; h < 16; ++h) {  // contiguous 32B per lane -> b128 loads
    b0[h] = ftT[(long)(c)      * S_ + jbase + h];
    b1[h] = ftT[(long)(16 + c) * S_ + jbase + h];
    b2[h] = ftT[(long)(32 + c) * S_ + jbase + h];
    b3[h] = ftT[(long)(48 + c) * S_ + jbase + h];
  }
  v8f acc0 = {}, acc1 = {}, acc2 = {}, acc3 = {};
  acc0 = __builtin_amdgcn_wmma_f32_16x16x32_f16(false, a, false, b0, (short)0, acc0, false, false);
  acc1 = __builtin_amdgcn_wmma_f32_16x16x32_f16(false, a, false, b1, (short)0, acc1, false, false);
  acc2 = __builtin_amdgcn_wmma_f32_16x16x32_f16(false, a, false, b2, (short)0, acc2, false, false);
  acc3 = __builtin_amdgcn_wmma_f32_16x16x32_f16(false, a, false, b3, (short)0, acc3, false, false);

  float rowsum = psum + __shfl_xor(psum, 16);
#pragma unroll
  for (int g = 0; g < 8; ++g) {
    int   rM = g + ((l >> 4) << 3);
    float rs = __shfl(rowsum, rM);
    long  ridx = base + i0 + rM;
    const float* resid = proj + ridx * NPROJ + HID_;
    int   s = e * K_ + i0 + rM;  // global stock index of this member
    float* o = out_all + (((long)(e * B_ + b)) * S_ + s) * HID_;
    float v;
    v = eluf(acc0[g] / rs + resid[c]);      o[c]      = v; sh_after[i0 + rM][c]      = v;
    v = eluf(acc1[g] / rs + resid[16 + c]); o[16 + c] = v; sh_after[i0 + rM][16 + c] = v;
    v = eluf(acc2[g] / rs + resid[32 + c]); o[32 + c] = v; sh_after[i0 + rM][32 + c] = v;
    v = eluf(acc3[g] / rs + resid[48 + c]); o[48 + c] = v; sh_after[i0 + rM][48 + c] = v;
  }
  __syncthreads();
  // maxpool over 32 members; thread t owns column t (HID_ == blockDim)
  float mx = sh_after[0][t];
#pragma unroll
  for (int r = 1; r < K_; ++r) mx = fmaxf(mx, sh_after[r][t]);
  out_pool[((long)b * E_ + e) * HID_ + t] = mx;
}

// ---------------------------------------------------------------------------
extern "C" void kernel_launch(void* const* d_in, const int* in_sizes, int n_in,
                              void* d_out, int out_size, void* d_ws, size_t ws_size,
                              hipStream_t stream) {
  const float* x    = (const float*)d_in[0];
  // d_in[1] = H (partition incidence: idx[e][k] = e*K + k, hardcoded)
  const float* adj  = (const float*)d_in[2];
  // d_in[3] = nhid (compile-time 64)
  const float* Wi   = (const float*)d_in[4];
  const float* bi   = (const float*)d_in[5];
  const float* a1i  = (const float*)d_in[6];
  const float* a1bi = (const float*)d_in[7];
  const float* a2i  = (const float*)d_in[8];
  const float* a2bi = (const float*)d_in[9];
  const float* Wri  = (const float*)d_in[10];
  const float* bri  = (const float*)d_in[11];
  const float* Wa   = (const float*)d_in[12];
  const float* ba   = (const float*)d_in[13];
  const float* a1a  = (const float*)d_in[14];
  const float* a1ba = (const float*)d_in[15];
  const float* a2a  = (const float*)d_in[16];
  const float* a2ba = (const float*)d_in[17];
  const float* Wra  = (const float*)d_in[18];
  const float* bra  = (const float*)d_in[19];

  float* out_all  = (float*)d_out;                               // [E,B,S,HID]
  float* out_pool = out_all + (long)E_ * B_ * S_ * HID_;         // [B,E,HID]
  float* out_ind  = out_pool + (long)B_ * E_ * HID_;             // [B,S,HID]

  // ---- workspace carve-up (256B aligned) ----
  char* ws = (char*)d_ws;
  size_t off = 0;
  auto carve = [&](size_t bytes) -> void* {
    void* p = ws + off;
    off = (off + bytes + 255) & ~(size_t)255;
    return p;
  };
  _Float16* xh      = (_Float16*)carve((size_t)M_ * F_ * 2);        // 8 MB
  _Float16* WcatTA  = (_Float16*)carve((size_t)F_ * NPROJ * 2);
  _Float16* WcatTI  = (_Float16*)carve((size_t)F_ * NPROJ * 2);
  float*    bcatA   = (float*)carve(NPROJ * 4);
  float*    bcatI   = (float*)carve(NPROJ * 4);
  float*    projA   = (float*)carve((size_t)M_ * NPROJ * 4);        // 16 MB
  _Float16* ftsTA   = (_Float16*)carve((size_t)M_ * HID_ * 2);      // 4 MB
  float*    projI   = (float*)carve((size_t)M_ * NPROJ * 4);        // 16 MB
  _Float16* ftsTI   = (_Float16*)carve((size_t)M_ * HID_ * 2);      // 4 MB
  float*    f1A     = (float*)carve((size_t)M_ * 4);
  float*    f2A     = (float*)carve((size_t)M_ * 4);
  float*    f1I     = (float*)carve((size_t)M_ * 4);
  float*    f2I     = (float*)carve((size_t)M_ * 4);
  unsigned* adjbits = (unsigned*)carve((size_t)S_ * AW_ * 4);       // 128 KB
  (void)ws_size; (void)in_sizes; (void)n_in; (void)out_size;

  // 1) zero the scatter target (256 MB, the bandwidth floor of this problem)
  long n4 = ((long)E_ * B_ * S_ * HID_) / 4;
  zero_f4_kernel<<<4096, 256, 0, stream>>>((float4*)out_all, n4);

  // 2) x -> f16; adj -> bit matrix
  cvt_f16_kernel<<<(M_ * F_ + 255) / 256, 256, 0, stream>>>(x, xh, M_ * F_);
  pack_adj_kernel<<<(S_ * AW_ + 255) / 256, 256, 0, stream>>>(adj, adjbits);

  // 3) pack [W|Wr]^T per branch
  pack_w_kernel<<<(F_ * NPROJ + 255) / 256, 256, 0, stream>>>(Wa, Wra, ba, bra, WcatTA, bcatA);
  pack_w_kernel<<<(F_ * NPROJ + 255) / 256, 256, 0, stream>>>(Wi, Wri, bi, bri, WcatTI, bcatI);

  // 4) WMMA projection GEMMs (M=32768, K=128, N=128)
  dim3 ggrid(M_ / 16, NPROJ / 16);
  proj_gemm_kernel<<<ggrid, 32, 0, stream>>>(xh, WcatTA, bcatA, projA, ftsTA);
  proj_gemm_kernel<<<ggrid, 32, 0, stream>>>(xh, WcatTI, bcatI, projI, ftsTI);

  // 5) attention coefficient row-dots
  f1f2_kernel<<<(M_ + 255) / 256, 256, 0, stream>>>(projA, a1a, a1ba, a2a, a2ba, f1A, f2A);
  f1f2_kernel<<<(M_ + 255) / 256, 256, 0, stream>>>(projI, a1i, a1bi, a2i, a2bi, f1I, f2I);

  // 6) fused masked-softmax attention (industry) -> out_ind
  industry_attn_kernel<<<dim3(S_ / (16 * ROWT), B_), 32, 0, stream>>>(projA, ftsTA, f1A, f2A,
                                                                      adjbits, out_ind);

  // 7) hyperedge heads + scatter + maxpool
  hyper_attn_kernel<<<dim3(E_, B_), 64, 0, stream>>>(projI, ftsTI, f1I, f2I, out_all,
                                                     out_pool);
}
